// LongFormer_7662221656596
// MI455X (gfx1250) — matrix operand order repaired and checked by reading, not compiled, not run on hardware
//
#include <hip/hip_runtime.h>
#include <hip/hip_bf16.h>
#include <math.h>
#include <stdint.h>

// ---------------- problem constants (match setup_inputs) ----------------
#define BB 2
#define SS 4096
#define DD 768
#define HH 12
#define DH 64
#define WW 128
#define GG 128
#define FF 3072
#define CC (SS / WW)   // 32 chunks
#define LL 2
#define NEGV (-1000000000.0f)

typedef __bf16 bf16_t;
typedef __attribute__((ext_vector_type(16))) __bf16 v16bf;
typedef __attribute__((ext_vector_type(8)))  float  v8f;
typedef int v4i_vec __attribute__((vector_size(16)));

// ---------------- async copy to LDS (gfx1250), guarded -------------------
#if defined(__has_builtin)
#  if __has_builtin(__builtin_amdgcn_global_load_async_to_lds_b128)
#    define HAVE_ASYNC 1
#  endif
#endif
#ifndef HAVE_ASYNC
#  define HAVE_ASYNC 0
#endif

__device__ __forceinline__ void cp16_async(void* lds, const void* g) {
#if HAVE_ASYNC
  __builtin_amdgcn_global_load_async_to_lds_b128(
      (__attribute__((address_space(1))) v4i_vec*)(uintptr_t)g,
      (__attribute__((address_space(3))) v4i_vec*)(unsigned)(uintptr_t)lds,
      0, 0);
#else
  *(uint4*)lds = *(const uint4*)g;
#endif
}
__device__ __forceinline__ void async_wait_all() {
#if HAVE_ASYNC
#  if __has_builtin(__builtin_amdgcn_s_wait_asynccnt)
  __builtin_amdgcn_s_wait_asynccnt(0);
#  else
  asm volatile("s_wait_asynccnt 0x0" ::: "memory");
#  endif
#endif
}

// fp32 -> bf16 round-to-nearest-even
__device__ __forceinline__ bf16_t f2bf(float f) {
  unsigned u = __builtin_bit_cast(unsigned, f);
  unsigned r = u + 0x7FFFu + ((u >> 16) & 1u);
  unsigned short hs = (unsigned short)(r >> 16);
  return __builtin_bit_cast(bf16_t, hs);
}
#define BF0 (__builtin_bit_cast(bf16_t, (unsigned short)0))

__device__ __forceinline__ float geluf(float x) {
  return 0.5f * x * (1.0f + erff(x * 0.70710678118654752f));
}

// ---------------- WMMA fragment helpers (CDNA5 16x16x32 bf16 layouts) ----
// A fragment: 16x32, tile stored [row][ldk] (K contiguous).
__device__ __forceinline__ v16bf frag_a_bf(const bf16_t* p0, int row, int ldk,
                                           int kofs, int lane) {
  int hlf = lane >> 4;
  int m   = lane & 15;
  const bf16_t* p = p0 + (row + m) * ldk + kofs + 8 * hlf;
  v16bf f;
#pragma unroll
  for (int j = 0; j < 8; ++j) {
    int kb = 2 * (j & 3) + 16 * (j >> 2);
    f[2 * j]     = p[kb];
    f[2 * j + 1] = p[kb + 1];
  }
  return f;
}

// A layout from an f32 source (softmax probs), convert on load.
__device__ __forceinline__ v16bf frag_a_f32(const float* p0, int row, int ldk,
                                            int kofs, int lane) {
  int hlf = lane >> 4;
  int m   = lane & 15;
  const float* p = p0 + (size_t)(row + m) * ldk + kofs + 8 * hlf;
  v16bf f;
#pragma unroll
  for (int j = 0; j < 8; ++j) {
    int kb = 2 * (j & 3) + 16 * (j >> 2);
    f[2 * j]     = f2bf(p[kb]);
    f[2 * j + 1] = f2bf(p[kb + 1]);
  }
  return f;
}

// B fragment: 32x16 (KxN), tile stored transposed [n][ldk].
__device__ __forceinline__ v16bf frag_b_bf(const bf16_t* p0, int ncol, int ldk,
                                           int kofs, int lane) {
  int hlf = lane >> 4;
  int n   = lane & 15;
  const bf16_t* p = p0 + (ncol + n) * ldk + kofs + 8 * hlf;
  v16bf f;
#pragma unroll
  for (int j = 0; j < 8; ++j) {
    int kb = 2 * (j & 3) + 16 * (j >> 2);
    f[2 * j]     = p[kb];
    f[2 * j + 1] = p[kb + 1];
  }
  return f;
}

__device__ __forceinline__ v8f wmma_bf16(v16bf a, v16bf b, v8f c) {
  return __builtin_amdgcn_wmma_f32_16x16x32_bf16(false, a, false, b,
                                                 (short)0, c, false, false);
}

// ---------------- block reductions (256 threads) -------------------------
__device__ __forceinline__ float blk_sum(float v, float* red) {
  int t = threadIdx.x;
  red[t] = v; __syncthreads();
#pragma unroll
  for (int off = 128; off > 0; off >>= 1) {
    if (t < off) red[t] += red[t + off];
    __syncthreads();
  }
  float r = red[0]; __syncthreads();
  return r;
}
__device__ __forceinline__ float blk_max(float v, float* red) {
  int t = threadIdx.x;
  red[t] = v; __syncthreads();
#pragma unroll
  for (int off = 128; off > 0; off >>= 1) {
    if (t < off) red[t] = fmaxf(red[t], red[t + off]);
    __syncthreads();
  }
  float r = red[0]; __syncthreads();
  return r;
}

// ---------------- weight convert + transpose: W(KxN f32) -> Wt(NxK bf16) -
__global__ __launch_bounds__(256) void wconv_t_kernel(
    const float* __restrict__ W, bf16_t* __restrict__ Wt, int K, int N) {
  __shared__ float tile[32][33];
  int bk = blockIdx.x * 32, bn = blockIdx.y * 32;
  int tx = threadIdx.x & 31, ty = threadIdx.x >> 5;  // 8 rows per pass
#pragma unroll
  for (int i = 0; i < 32; i += 8)
    tile[ty + i][tx] = W[(size_t)(bk + ty + i) * N + bn + tx];
  __syncthreads();
#pragma unroll
  for (int i = 0; i < 32; i += 8)
    Wt[(size_t)(bn + ty + i) * K + bk + tx] = f2bf(tile[tx][ty + i]);
}

// ---------------- embedding + LayerNorm (writes f32 + bf16) --------------
__global__ __launch_bounds__(256) void embed_ln_kernel(
    const int* __restrict__ x, const float* __restrict__ ew,
    const float* __restrict__ ep, const float* __restrict__ g,
    const float* __restrict__ bta, float* __restrict__ h,
    bf16_t* __restrict__ hb) {
  __shared__ float red[256];
  int bs = blockIdx.x;
  int s  = bs % SS;
  int tok = x[bs];
  float vals[3]; float lsum = 0.f;
#pragma unroll
  for (int i = 0; i < 3; ++i) {
    int d = threadIdx.x + i * 256;
    vals[i] = ew[(size_t)tok * DD + d] + ep[(size_t)s * DD + d];
    lsum += vals[i];
  }
  float mean = blk_sum(lsum, red) * (1.0f / DD);
  float lvar = 0.f;
#pragma unroll
  for (int i = 0; i < 3; ++i) { float d = vals[i] - mean; lvar += d * d; }
  float var = blk_sum(lvar, red) * (1.0f / DD);
  float rstd = rsqrtf(var + 1e-5f);
#pragma unroll
  for (int i = 0; i < 3; ++i) {
    int d = threadIdx.x + i * 256;
    float o = (vals[i] - mean) * rstd * g[d] + bta[d];
    h[(size_t)bs * DD + d] = o;
    hb[(size_t)bs * DD + d] = f2bf(o);
  }
}

// ---------------- residual add + LayerNorm (in place, writes bf16 too) ---
__global__ __launch_bounds__(256) void add_ln_kernel(
    float* __restrict__ h, const float* __restrict__ t,
    const float* __restrict__ g, const float* __restrict__ bta,
    bf16_t* __restrict__ hb) {
  __shared__ float red[256];
  int bs = blockIdx.x;
  float vals[3]; float lsum = 0.f;
#pragma unroll
  for (int i = 0; i < 3; ++i) {
    int d = threadIdx.x + i * 256;
    vals[i] = h[(size_t)bs * DD + d] + t[(size_t)bs * DD + d];
    lsum += vals[i];
  }
  float mean = blk_sum(lsum, red) * (1.0f / DD);
  float lvar = 0.f;
#pragma unroll
  for (int i = 0; i < 3; ++i) { float d = vals[i] - mean; lvar += d * d; }
  float var = blk_sum(lvar, red) * (1.0f / DD);
  float rstd = rsqrtf(var + 1e-5f);
#pragma unroll
  for (int i = 0; i < 3; ++i) {
    int d = threadIdx.x + i * 256;
    float o = (vals[i] - mean) * rstd * g[d] + bta[d];
    h[(size_t)bs * DD + d] = o;
    hb[(size_t)bs * DD + d] = f2bf(o);
  }
}

// ---------------- bf16 WMMA GEMM, 128x128 tile, double-buffered LDS ------
// A: MxK bf16 row-major.  Bt: NxK bf16 (pre-transposed).  Outputs:
// C (f32, optional) and/or Cb (bf16, optional) = act((A*B + bias) * scale).
__global__ __launch_bounds__(256) void gemm_bf16_kernel(
    const bf16_t* __restrict__ A, const bf16_t* __restrict__ Bt,
    const float* __restrict__ bias, float* __restrict__ C,
    bf16_t* __restrict__ Cb, int M, int N, int K, float scale, int act) {
  __shared__ __align__(16) bf16_t As[2][128][32];
  __shared__ __align__(16) bf16_t Bs[2][128][32];
  int tid = threadIdx.x, lane = tid & 31, wave = tid >> 5;
  int bm = blockIdx.x * 128, bn = blockIdx.y * 128;
  int wm = (wave >> 1) * 32, wn = (wave & 1) * 64;
  v8f acc[2][4];
#pragma unroll
  for (int i = 0; i < 2; ++i)
#pragma unroll
    for (int j = 0; j < 4; ++j) acc[i][j] = (v8f){};

  auto stage = [&](int buf, int k0) {
#pragma unroll
    for (int i = 0; i < 2; ++i) {   // A: 128 rows x 4 x 16B
      int u = tid + i * 256;
      int r = u >> 2, seg = (u & 3) * 8;
      cp16_async(&As[buf][r][seg], A + (size_t)(bm + r) * K + k0 + seg);
    }
#pragma unroll
    for (int i = 0; i < 2; ++i) {   // B: 128 rows x 4 x 16B
      int u = tid + i * 256;
      int r = u >> 2, seg = (u & 3) * 8;
      cp16_async(&Bs[buf][r][seg], Bt + (size_t)(bn + r) * K + k0 + seg);
    }
  };

  int nk = K / 32;
  stage(0, 0);
  for (int kb = 0; kb < nk; ++kb) {
    async_wait_all();
    __syncthreads();
    if (kb + 1 < nk) stage((kb + 1) & 1, (kb + 1) * 32);
    int cur = kb & 1;
    v16bf a0 = frag_a_bf(&As[cur][0][0], wm, 32, 0, lane);
    v16bf a1 = frag_a_bf(&As[cur][0][0], wm + 16, 32, 0, lane);
#pragma unroll
    for (int nt = 0; nt < 4; ++nt) {
      v16bf bfr = frag_b_bf(&Bs[cur][0][0], wn + nt * 16, 32, 0, lane);
      acc[0][nt] = wmma_bf16(a0, bfr, acc[0][nt]);
      acc[1][nt] = wmma_bf16(a1, bfr, acc[1][nt]);
    }
  }

  int hlf = lane >> 4, nl = lane & 15;
#pragma unroll
  for (int mt = 0; mt < 2; ++mt)
#pragma unroll
    for (int nt = 0; nt < 4; ++nt)
#pragma unroll
      for (int r = 0; r < 8; ++r) {
        int row = bm + wm + mt * 16 + r + 8 * hlf;
        int col = bn + wn + nt * 16 + nl;
        float vv = (acc[mt][nt][r] + bias[col]) * scale;
        if (act) vv = geluf(vv);
        if (C)  C[(size_t)row * N + col] = vv;
        if (Cb) Cb[(size_t)row * N + col] = f2bf(vv);
      }
}

// ---------------- banded local attention (bf16 q/k/v, bf16 out) ----------
__global__ __launch_bounds__(256) void band_attn_kernel(
    const bf16_t* __restrict__ q, const bf16_t* __restrict__ k,
    const bf16_t* __restrict__ v, const int* __restrict__ mask,
    bf16_t* __restrict__ attnb) {
  __shared__ __align__(16) bf16_t Qs[16][64];
  __shared__ __align__(16) bf16_t Ks[8][16][64];
  __shared__ float  Ss[16][512];
  __shared__ bf16_t Vs[64][32];
  int idx = blockIdx.x;
  int qb = idx & 7;   idx >>= 3;
  int c  = idx % CC;  idx /= CC;
  int hh = idx % HH;  idx /= HH;
  int b  = idx;
  int tid = threadIdx.x, lane = tid & 31, wave = tid >> 5;
  int hlf = lane >> 4, nl = lane & 15;

  // stage 16x64 query tile (16B units)
  if (tid < 128) {
    int m = tid >> 3, seg = (tid & 7) * 8;
    int qrow = c * WW + qb * 16 + m;
    cp16_async(&Qs[m][seg], q + ((size_t)b * SS + qrow) * DD + hh * DH + seg);
  }
  async_wait_all();
  __syncthreads();

  // ---- scores: 512 columns = 32 n-tiles, 8 per pass ----
  for (int kt = 0; kt < 4; ++kt) {
#pragma unroll
    for (int i = 0; i < 4; ++i) {   // 1024 16B units
      int u = tid + i * 256;
      int sw = u >> 7, rem = u & 127;
      int j = rem >> 3, seg = (rem & 7) * 8;
      int n = (kt * 8 + sw) * 16 + j;
      int krow;
      if (n < GG) krow = n;
      else {
        int kabs = c * WW + (n - GG) - WW;
        krow = (kabs >= 0 && kabs < SS) ? kabs : -1;
      }
      if (krow >= 0)
        cp16_async(&Ks[sw][j][seg],
                   k + ((size_t)b * SS + krow) * DD + hh * DH + seg);
      else
        *(uint4*)&Ks[sw][j][seg] = (uint4){0u, 0u, 0u, 0u};
    }
    async_wait_all();
    __syncthreads();
    v8f sacc = {};
    v16bf a0 = frag_a_bf(&Qs[0][0], 0, 64, 0, lane);
    v16bf b0 = frag_b_bf(&Ks[wave][0][0], 0, 64, 0, lane);
    sacc = wmma_bf16(a0, b0, sacc);
    v16bf a1 = frag_a_bf(&Qs[0][0], 0, 64, 32, lane);
    v16bf b1 = frag_b_bf(&Ks[wave][0][0], 0, 64, 32, lane);
    sacc = wmma_bf16(a1, b1, sacc);
    int n = (kt * 8 + wave) * 16 + nl;
#pragma unroll
    for (int r = 0; r < 8; ++r) {
      int m = r + 8 * hlf;
      int qpos = c * WW + qb * 16 + m;
      bool ok;
      if (n < GG) ok = true;  // global columns unmasked pre-softmax
      else {
        int kabs = c * WW + (n - GG) - WW;
        int rel = kabs - qpos;
        ok = (kabs >= 0) && (kabs < SS) && (kabs >= GG) &&
             (rel >= -WW) && (rel <= WW) && (mask[b * SS + kabs] > 0);
      }
      Ss[m][n] = ok ? sacc[r] : NEGV;
    }
    __syncthreads();
  }

  // ---- softmax over 512 columns, 16 threads per row ----
  {
    int row = tid >> 4, i16 = tid & 15;
    float mx = -3.4e38f;
    for (int col = i16; col < 512; col += 16) mx = fmaxf(mx, Ss[row][col]);
#pragma unroll
    for (int off = 8; off > 0; off >>= 1) mx = fmaxf(mx, __shfl_xor(mx, off, 32));
    float sm = 0.f;
    for (int col = i16; col < 512; col += 16) sm += __expf(Ss[row][col] - mx);
#pragma unroll
    for (int off = 8; off > 0; off >>= 1) sm += __shfl_xor(sm, off, 32);
    float inv = 1.0f / sm;
    for (int col = i16; col < 512; col += 16)
      Ss[row][col] = __expf(Ss[row][col] - mx) * inv;
  }
  __syncthreads();

  // ---- PV: out(16x64) = probs(16x512) x V(512x64); waves 0..3 own ntiles
  v8f oacc = {};
  for (int ks = 0; ks < 16; ++ks) {
    // transpose-stage V (scalar: [d][kk] from [key][d])
#pragma unroll
    for (int i = 0; i < 8; ++i) {
      int e = tid + i * 256;
      int d = e >> 5, kk = e & 31;
      int n = ks * 32 + kk;
      int vrow;
      if (n < GG) vrow = n;
      else {
        int kabs = c * WW + (n - GG) - WW;
        vrow = (kabs >= 0 && kabs < SS) ? kabs : -1;
      }
      Vs[d][kk] = (vrow >= 0)
          ? v[((size_t)b * SS + vrow) * DD + hh * DH + d] : BF0;
    }
    __syncthreads();
    if (wave < 4) {
      v16bf a = frag_a_f32(&Ss[0][0], 0, 512, ks * 32, lane);
      v16bf bb = frag_b_bf(&Vs[0][0], wave * 16, 32, 0, lane);
      oacc = wmma_bf16(a, bb, oacc);
    }
    __syncthreads();
  }
  if (wave < 4) {
#pragma unroll
    for (int r = 0; r < 8; ++r) {
      int m = r + 8 * hlf;
      int srow = c * WW + qb * 16 + m;
      attnb[((size_t)b * SS + srow) * DD + hh * DH + wave * 16 + nl] =
          f2bf(oacc[r]);
    }
  }
}

// ---------------- global attention: scores -------------------------------
__global__ __launch_bounds__(256) void glob_scores_kernel(
    const bf16_t* __restrict__ qg, const bf16_t* __restrict__ kg,
    const int* __restrict__ mask, float* __restrict__ SG) {
  __shared__ __align__(16) bf16_t Qs[128][64];
  __shared__ __align__(16) bf16_t Ks[128][64];
  int sblk = blockIdx.x;
  int bh = blockIdx.y;
  int hh = bh % HH, b = bh / HH;
  int tid = threadIdx.x, lane = tid & 31, wave = tid >> 5;
#pragma unroll
  for (int i = 0; i < 4; ++i) {   // 1024 units each
    int u = tid + i * 256;
    int r = u >> 3, seg = (u & 7) * 8;
    cp16_async(&Qs[r][seg], qg + ((size_t)b * SS + r) * DD + hh * DH + seg);
    cp16_async(&Ks[r][seg],
               kg + ((size_t)b * SS + sblk * 128 + r) * DD + hh * DH + seg);
  }
  async_wait_all();
  __syncthreads();
  int wm = wave * 16;
  int hlf = lane >> 4, nl = lane & 15;
  v16bf a0 = frag_a_bf(&Qs[0][0], wm, 64, 0, lane);
  v16bf a1 = frag_a_bf(&Qs[0][0], wm, 64, 32, lane);
#pragma unroll
  for (int nt = 0; nt < 8; ++nt) {
    v8f acc = {};
    v16bf b0 = frag_b_bf(&Ks[0][0], nt * 16, 64, 0, lane);
    v16bf b1 = frag_b_bf(&Ks[0][0], nt * 16, 64, 32, lane);
    acc = wmma_bf16(a0, b0, acc);
    acc = wmma_bf16(a1, b1, acc);
    int sc = sblk * 128 + nt * 16 + nl;
    bool ok = mask[b * SS + sc] > 0;
#pragma unroll
    for (int r = 0; r < 8; ++r) {
      int grow = wm + r + 8 * hlf;
      SG[(((size_t)b * HH + hh) * GG + grow) * SS + sc] = ok ? acc[r] : NEGV;
    }
  }
}

// ---------------- global attention: row softmax over S -------------------
__global__ __launch_bounds__(256) void glob_softmax_kernel(float* __restrict__ SG) {
  __shared__ float red[256];
  size_t row = blockIdx.x;
  float* p = SG + row * SS;
  float mx = -3.4e38f;
  for (int c = threadIdx.x; c < SS; c += 256) mx = fmaxf(mx, p[c]);
  mx = blk_max(mx, red);
  float sm = 0.f;
  for (int c = threadIdx.x; c < SS; c += 256) sm += __expf(p[c] - mx);
  sm = blk_sum(sm, red);
  float inv = 1.0f / sm;
  for (int c = threadIdx.x; c < SS; c += 256) p[c] = __expf(p[c] - mx) * inv;
}

// ---------------- global attention: PV, overwrite rows < G ---------------
__global__ __launch_bounds__(256) void glob_pv_kernel(
    const float* __restrict__ SG, const bf16_t* __restrict__ vg,
    bf16_t* __restrict__ attnb) {
  __shared__ bf16_t Vs[64][32];
  int bh = blockIdx.x;
  int hh = bh % HH, b = bh / HH;
  int tid = threadIdx.x, lane = tid & 31, wave = tid >> 5;
  int hlf = lane >> 4, nl = lane & 15;
  int mrow = wave * 16;
  const float* prow0 = SG + (((size_t)b * HH + hh) * GG + mrow) * SS;
  v8f acc[4] = {{}, {}, {}, {}};
  for (int ks = 0; ks < SS / 32; ++ks) {
#pragma unroll
    for (int i = 0; i < 8; ++i) {   // transpose-stage (scalar)
      int e = tid + i * 256;
      int d = e >> 5, kk = e & 31;
      Vs[d][kk] = vg[((size_t)b * SS + ks * 32 + kk) * DD + hh * DH + d];
    }
    __syncthreads();
    v16bf a = frag_a_f32(prow0, 0, SS, ks * 32, lane);
#pragma unroll
    for (int nt = 0; nt < 4; ++nt) {
      v16bf bb = frag_b_bf(&Vs[0][0], nt * 16, 32, 0, lane);
      acc[nt] = wmma_bf16(a, bb, acc[nt]);
    }
    __syncthreads();
  }
#pragma unroll
  for (int nt = 0; nt < 4; ++nt)
#pragma unroll
    for (int r = 0; r < 8; ++r) {
      int grow = mrow + r + 8 * hlf;   // < G
      attnb[((size_t)b * SS + grow) * DD + hh * DH + nt * 16 + nl] =
          f2bf(acc[nt][r]);
    }
}

// ---------------- classifier head ----------------------------------------
__global__ void cls1_kernel(const float* __restrict__ h,
                            const float* __restrict__ W,
                            const float* __restrict__ bias,
                            float* __restrict__ out) {
  int idx = blockIdx.x * blockDim.x + threadIdx.x;
  if (idx >= BB * DD) return;
  int b = idx / DD, j = idx % DD;
  const float* hr = h + (size_t)b * SS * DD;  // row s=0
  float acc = bias[j];
  for (int d = 0; d < DD; ++d) acc += hr[d] * W[(size_t)d * DD + j];
  out[idx] = geluf(acc);
}
__global__ __launch_bounds__(256) void cls2_kernel(
    const float* __restrict__ m1, const float* __restrict__ W2,
    const float* __restrict__ b2, float* __restrict__ out) {
  __shared__ float red[256];
  int b = blockIdx.x;
  float acc = 0.f;
  for (int j = threadIdx.x; j < DD; j += 256) acc += m1[b * DD + j] * W2[j];
  acc = blk_sum(acc, red);
  if (threadIdx.x == 0) out[b] = acc + b2[0];
}

// ---------------- host orchestration -------------------------------------
extern "C" void kernel_launch(void* const* d_in, const int* in_sizes, int n_in,
                              void* d_out, int out_size, void* d_ws,
                              size_t ws_size, hipStream_t stream) {
  (void)in_sizes; (void)n_in; (void)out_size; (void)ws_size;
  const int*   x        = (const int*)d_in[0];
  const int*   maskp    = (const int*)d_in[1];
  const float* emb_word = (const float*)d_in[4];
  const float* emb_pos  = (const float*)d_in[5];
  const float* eln_g    = (const float*)d_in[6];
  const float* eln_b    = (const float*)d_in[7];
  const float* Wq  = (const float*)d_in[8];  const float* bq  = (const float*)d_in[9];
  const float* Wk  = (const float*)d_in[10]; const float* bk  = (const float*)d_in[11];
  const float* Wv  = (const float*)d_in[12]; const float* bv  = (const float*)d_in[13];
  const float* Wqg = (const float*)d_in[14]; const float* bqg = (const float*)d_in[15];
  const float* Wkg = (const float*)d_in[16]; const float* bkg = (const float*)d_in[17];
  const float* Wvg = (const float*)d_in[18]; const float* bvg = (const float*)d_in[19];
  const float* Wo  = (const float*)d_in[20]; const float* bo  = (const float*)d_in[21];
  const float* l1g = (const float*)d_in[22]; const float* l1b = (const float*)d_in[23];
  const float* W1  = (const float*)d_in[24]; const float* b1  = (const float*)d_in[25];
  const float* W2  = (const float*)d_in[26]; const float* b2  = (const float*)d_in[27];
  const float* l2g = (const float*)d_in[28]; const float* l2b = (const float*)d_in[29];
  const float* cW1 = (const float*)d_in[30]; const float* cb1 = (const float*)d_in[31];
  const float* cW2 = (const float*)d_in[32]; const float* cb2 = (const float*)d_in[33];

  char* base = (char*)d_ws;
  auto alloc = [&](size_t bytes) {
    char* p = base;
    base += (bytes + 255) & ~(size_t)255;
    return (void*)p;
  };
  size_t BSD = (size_t)BB * SS * DD;
  float*  h     = (float*)alloc(BSD * 4);
  float*  tmp   = (float*)alloc(BSD * 4);
  float*  SG    = (float*)alloc((size_t)BB * HH * GG * SS * 4);
  float*  clsm  = (float*)alloc((size_t)BB * DD * 4);
  bf16_t* hb    = (bf16_t*)alloc(BSD * 2);
  bf16_t* qb_   = (bf16_t*)alloc(BSD * 2);
  bf16_t* kb_   = (bf16_t*)alloc(BSD * 2);
  bf16_t* vb_   = (bf16_t*)alloc(BSD * 2);
  bf16_t* qgb   = (bf16_t*)alloc(BSD * 2);
  bf16_t* kgb   = (bf16_t*)alloc(BSD * 2);
  bf16_t* vgb   = (bf16_t*)alloc(BSD * 2);
  bf16_t* attnb = (bf16_t*)alloc(BSD * 2);
  bf16_t* midb  = (bf16_t*)alloc((size_t)BB * SS * FF * 2);
  bf16_t* wt    = (bf16_t*)alloc((size_t)DD * FF * 2);

  const int BS = BB * SS;          // 8192 rows
  const float qscale = 0.125f;     // 1/sqrt(64)
  dim3 gDD(BS / 128, DD / 128);    // GEMM grids (128x128 tiles)
  dim3 gDF(BS / 128, FF / 128);
  dim3 wDD(DD / 32, DD / 32);      // transpose grids (K/32, N/32)
  dim3 wDF(DD / 32, FF / 32);
  dim3 wFD(FF / 32, DD / 32);

  embed_ln_kernel<<<BB * SS, 256, 0, stream>>>(x, emb_word, emb_pos, eln_g,
                                               eln_b, h, hb);

  for (int l = 0; l < LL; ++l) {
    const float* Wq_ = Wq + (size_t)l * DD * DD;  const float* bq_ = bq + (size_t)l * DD;
    const float* Wk_ = Wk + (size_t)l * DD * DD;  const float* bk_ = bk + (size_t)l * DD;
    const float* Wv_ = Wv + (size_t)l * DD * DD;  const float* bv_ = bv + (size_t)l * DD;
    const float* Wqg_ = Wqg + (size_t)l * DD * DD; const float* bqg_ = bqg + (size_t)l * DD;
    const float* Wkg_ = Wkg + (size_t)l * DD * DD; const float* bkg_ = bkg + (size_t)l * DD;
    const float* Wvg_ = Wvg + (size_t)l * DD * DD; const float* bvg_ = bvg + (size_t)l * DD;
    const float* Wo_ = Wo + (size_t)l * DD * DD;  const float* bo_ = bo + (size_t)l * DD;
    const float* W1_ = W1 + (size_t)l * DD * FF;  const float* b1_ = b1 + (size_t)l * FF;
    const float* W2_ = W2 + (size_t)l * FF * DD;  const float* b2_ = b2 + (size_t)l * DD;

    wconv_t_kernel<<<wDD, 256, 0, stream>>>(Wq_, wt, DD, DD);
    gemm_bf16_kernel<<<gDD, 256, 0, stream>>>(hb, wt, bq_, nullptr, qb_,
                                              BS, DD, DD, qscale, 0);
    wconv_t_kernel<<<wDD, 256, 0, stream>>>(Wk_, wt, DD, DD);
    gemm_bf16_kernel<<<gDD, 256, 0, stream>>>(hb, wt, bk_, nullptr, kb_,
                                              BS, DD, DD, 1.0f, 0);
    wconv_t_kernel<<<wDD, 256, 0, stream>>>(Wv_, wt, DD, DD);
    gemm_bf16_kernel<<<gDD, 256, 0, stream>>>(hb, wt, bv_, nullptr, vb_,
                                              BS, DD, DD, 1.0f, 0);
    wconv_t_kernel<<<wDD, 256, 0, stream>>>(Wqg_, wt, DD, DD);
    gemm_bf16_kernel<<<gDD, 256, 0, stream>>>(hb, wt, bqg_, nullptr, qgb,
                                              BS, DD, DD, qscale, 0);
    wconv_t_kernel<<<wDD, 256, 0, stream>>>(Wkg_, wt, DD, DD);
    gemm_bf16_kernel<<<gDD, 256, 0, stream>>>(hb, wt, bkg_, nullptr, kgb,
                                              BS, DD, DD, 1.0f, 0);
    wconv_t_kernel<<<wDD, 256, 0, stream>>>(Wvg_, wt, DD, DD);
    gemm_bf16_kernel<<<gDD, 256, 0, stream>>>(hb, wt, bvg_, nullptr, vgb,
                                              BS, DD, DD, 1.0f, 0);

    band_attn_kernel<<<BB * HH * CC * 8, 256, 0, stream>>>(qb_, kb_, vb_,
                                                           maskp, attnb);
    glob_scores_kernel<<<dim3(SS / 128, BB * HH), 256, 0, stream>>>(qgb, kgb,
                                                                    maskp, SG);
    glob_softmax_kernel<<<BB * HH * GG, 256, 0, stream>>>(SG);
    glob_pv_kernel<<<BB * HH, 256, 0, stream>>>(SG, vgb, attnb);

    wconv_t_kernel<<<wDD, 256, 0, stream>>>(Wo_, wt, DD, DD);
    gemm_bf16_kernel<<<gDD, 256, 0, stream>>>(attnb, wt, bo_, tmp, nullptr,
                                              BS, DD, DD, 1.0f, 0);
    add_ln_kernel<<<BB * SS, 256, 0, stream>>>(h, tmp, l1g + (size_t)l * DD,
                                               l1b + (size_t)l * DD, hb);

    wconv_t_kernel<<<wDF, 256, 0, stream>>>(W1_, wt, DD, FF);
    gemm_bf16_kernel<<<gDF, 256, 0, stream>>>(hb, wt, b1_, nullptr, midb,
                                              BS, FF, DD, 1.0f, 1);
    wconv_t_kernel<<<wFD, 256, 0, stream>>>(W2_, wt, FF, DD);
    gemm_bf16_kernel<<<gDD, 256, 0, stream>>>(midb, wt, b2_, tmp, nullptr,
                                              BS, DD, FF, 1.0f, 0);
    add_ln_kernel<<<BB * SS, 256, 0, stream>>>(h, tmp, l2g + (size_t)l * DD,
                                               l2b + (size_t)l * DD, hb);
  }

  cls1_kernel<<<(BB * DD + 255) / 256, 256, 0, stream>>>(h, cW1, cb1, clsm);
  cls2_kernel<<<BB, 256, 0, stream>>>(clsm, cW2, cb2, (float*)d_out);
}
